// MultiHeadAttention_10514079941230
// MI455X (gfx1250) — compile-verified, matrix operand order checked by reading, not run on hardware
//
#include <hip/hip_runtime.h>

typedef __attribute__((ext_vector_type(16))) __bf16 v16bf;
typedef __attribute__((ext_vector_type(8)))  __bf16 v8bf;
typedef __attribute__((ext_vector_type(8)))  float  v8f;
typedef __attribute__((ext_vector_type(4)))  unsigned int v4u;
typedef __attribute__((ext_vector_type(8)))  int v8i;
typedef __attribute__((ext_vector_type(4)))  int v4i;

#define D_MODEL 1024
#define SEQ     2048
#define BATCH   4
#define HEADS   16
#define HD      64
#define MTOT    (BATCH*SEQ)   // 8192

#define WMMA_BF16(a,b,c) \
  __builtin_amdgcn_wmma_f32_16x16x32_bf16(false,(a),false,(b),(short)0,(c),false,false)

__device__ inline v8f vzero8() {
  v8f z;
#pragma unroll
  for (int i = 0; i < 8; ++i) z[i] = 0.0f;
  return z;
}

__device__ inline v8bf cvt_f32x8_bf16(const float* p) {
  float4 f0 = *(const float4*)p;
  float4 f1 = *(const float4*)(p + 4);
  v8bf o;
  o[0]=(__bf16)f0.x; o[1]=(__bf16)f0.y; o[2]=(__bf16)f0.z; o[3]=(__bf16)f0.w;
  o[4]=(__bf16)f1.x; o[5]=(__bf16)f1.y; o[6]=(__bf16)f1.z; o[7]=(__bf16)f1.w;
  return o;
}

// A/B fragment gather for 16x16x32 bf16 WMMA: lanes 0-15 hold K {0..7,16..23},
// lanes 16-31 hold K {8..15,24..31}: two 16B chunks at +0 and +16 elements.
__device__ inline v16bf load_frag_ptr(const __bf16* p) {
  union { v16bf v; v8bf h[2]; } u;
  u.h[0] = *(const v8bf*)p;
  u.h[1] = *(const v8bf*)(p + 16);
  return u.v;
}
__device__ inline v16bf load_frag_lds(const __bf16* base, int ld, int row0, int col0, int lane) {
  int r = lane & 15, half = lane >> 4;
  return load_frag_ptr(base + (size_t)(row0 + r) * ld + col0 + half * 8);
}

// ---- DPP16 butterfly reductions over the 16-lane half (VALU only, no LDS) ----
template<int CTRL>
__device__ inline float dpp_xchg(float x) {
  int r = __builtin_amdgcn_update_dpp(0, __builtin_bit_cast(int, x), CTRL, 0xf, 0xf, true);
  return __builtin_bit_cast(float, r);
}
__device__ inline float red16_max(float v) {
  v = fmaxf(v, dpp_xchg<0x0B1>(v));   // quad_perm [1,0,3,2]  (xor 1)
  v = fmaxf(v, dpp_xchg<0x04E>(v));   // quad_perm [2,3,0,1]  (xor 2)
  v = fmaxf(v, dpp_xchg<0x141>(v));   // row_half_mirror      (8-group)
  v = fmaxf(v, dpp_xchg<0x140>(v));   // row_mirror           (16-group)
  return v;
}
__device__ inline float red16_sum(float v) {
  v += dpp_xchg<0x0B1>(v);
  v += dpp_xchg<0x04E>(v);
  v += dpp_xchg<0x141>(v);
  v += dpp_xchg<0x140>(v);
  return v;
}

// ---- Tensor Data Mover: 2D bf16 tile global->LDS with DMA-side LDS padding ----
// D# per CDNA5 ISA ch.8: group0 {count=1, lds_addr, global_addr(57b), type=2},
// group1 {data_size=2B, pad_enable, pad_interval, pad_amount, dims, tile, stride}.
__device__ inline void tdm_load_2d_bf16(unsigned lds_off, const void* gaddr,
                                        unsigned tensor_d0, unsigned tensor_d1,
                                        unsigned stride0,
                                        unsigned tile_d0, unsigned tile_d1,
                                        unsigned pad_int_code, unsigned pad_amt_code) {
  unsigned long long ga = (unsigned long long)gaddr;
  v4u g0;
  g0[0] = 1u;                                             // count=1
  g0[1] = lds_off;
  g0[2] = (unsigned)ga;
  g0[3] = (unsigned)((ga >> 32) & 0x01ffffffull) | (2u << 30);  // type=2 (image)
  v8i g1;
  g1[0] = (int)((1u << 16) | (1u << 20) |                 // data_size=2B, pad_enable
                (pad_int_code << 22) | (pad_amt_code << 25));
  g1[1] = (int)((tensor_d0 & 0xffffu) << 16);             // tensor_dim0[15:0] @bit48
  g1[2] = (int)((tensor_d0 >> 16) | ((tensor_d1 & 0xffffu) << 16));
  g1[3] = (int)((tensor_d1 >> 16) | (tile_d0 << 16));     // tile_dim0 @bit112
  g1[4] = (int)tile_d1;                                   // tile_dim1, tile_dim2=0
  g1[5] = (int)stride0;                                   // tensor_dim0_stride[31:0]
  g1[6] = 0; g1[7] = 0;                                   // stride hi, dim1_stride=0
  v4i gz; gz[0]=0; gz[1]=0; gz[2]=0; gz[3]=0;
#if __clang_major__ >= 23
  v8i gz8;
#pragma unroll
  for (int i = 0; i < 8; ++i) gz8[i] = 0;
  __builtin_amdgcn_tensor_load_to_lds(g0, g1, gz, gz, gz8, 0);
#else
  __builtin_amdgcn_tensor_load_to_lds(g0, g1, gz, gz, 0);
#endif
}
__device__ inline unsigned lds_addr_of(const void* p) {
  return (unsigned)(unsigned long long)p;   // low 32 bits of flat addr = LDS offset
}

// ---------------------------------------------------------------------------
// Shared epilogue: scatter C tile.
// MODE 0: bf16 [B,H,S,64]   MODE 1: bf16 [B,H,64,S]   MODE 2: f32 [M,1024]
// ---------------------------------------------------------------------------
template<int MODE>
__device__ inline void gemm_epilogue(v8f acc[4][2], const float* bias, void* Outp,
                                     int m0, int n0, int wm, int wn, int lane) {
  const int nl = lane & 15;
  const int mh = (lane >> 4) * 8;
#pragma unroll
  for (int mt = 0; mt < 4; ++mt)
#pragma unroll
    for (int nt = 0; nt < 2; ++nt)
#pragma unroll
      for (int i = 0; i < 8; ++i) {
        const int m = m0 + wm * 64 + mt * 16 + mh + i;
        const int n = n0 + wn * 32 + nt * 16 + nl;
        const float v = acc[mt][nt][i] + bias[n];
        if (MODE == 2) {
          ((float*)Outp)[(size_t)m * D_MODEL + n] = v;
        } else {
          const int b = m >> 11, s = m & (SEQ - 1);
          const int h = n >> 6,  d = n & 63;
          size_t idx;
          if (MODE == 0) idx = (((size_t)(b * HEADS + h)) * SEQ + s) * HD + d;
          else           idx = (((size_t)(b * HEADS + h)) * HD + d) * SEQ + s;
          ((__bf16*)Outp)[idx] = (__bf16)v;
        }
      }
}

// ---------------------------------------------------------------------------
// f32 -> bf16 bulk convert (bandwidth trivial; working set then lives in L2)
// ---------------------------------------------------------------------------
__global__ __launch_bounds__(256)
void cvt_f32_to_bf16(const float* __restrict__ in, __bf16* __restrict__ out, int n8) {
  const int i = blockIdx.x * 256 + threadIdx.x;
  if (i < n8) *(v8bf*)(out + (size_t)i * 8) = cvt_f32x8_bf16(in + (size_t)i * 8);
}

// ---------------------------------------------------------------------------
// GEMM v2: all-bf16 inputs, TDM double-buffered staging, BK=64.
// Out = A @ W^T + bias. Tile 128x128x64, 8 waves (2x4), wave tile 64x32.
// TDM pads each 32-dword row with 4 dwords -> LDS stride 72 elems.
// 16 WMMA per wave between barriers; 16 K-stages.
// ---------------------------------------------------------------------------
template<int MODE>
__global__ __launch_bounds__(256)
void gemm_tdm_wmma(const __bf16* __restrict__ Abf, const __bf16* __restrict__ Wbf,
                   const float* __restrict__ bias, void* __restrict__ Outp,
                   int a_rows) {
  constexpr int BK = 64, LDT = 72, NIT = D_MODEL / BK;
  __shared__ __bf16 As[2][128 * LDT];
  __shared__ __bf16 Bs[2][128 * LDT];

  const int tid  = threadIdx.x;
  const int lane = tid & 31;
  const int wav  = tid >> 5;
  const int wm   = wav & 1;
  const int wn   = wav >> 1;
  const int m0   = blockIdx.x * 128;
  const int n0   = blockIdx.y * 128;

  v8f acc[4][2];
#pragma unroll
  for (int mt = 0; mt < 4; ++mt) { acc[mt][0] = vzero8(); acc[mt][1] = vzero8(); }

  if (wav == 0) {
    tdm_load_2d_bf16(lds_addr_of(&As[0][0]), Abf + (size_t)m0 * D_MODEL,
                     D_MODEL, (unsigned)a_rows, D_MODEL, BK, 128, 4, 3);
    tdm_load_2d_bf16(lds_addr_of(&Bs[0][0]), Wbf + (size_t)n0 * D_MODEL,
                     D_MODEL, D_MODEL, D_MODEL, BK, 128, 4, 3);
  }

  for (int it = 0; it < NIT; ++it) {
    const int cur = it & 1;
    if (wav == 0) {
      if (it + 1 < NIT) {
        const int kt = (it + 1) * BK;
        tdm_load_2d_bf16(lds_addr_of(&As[cur ^ 1][0]), Abf + (size_t)m0 * D_MODEL + kt,
                         D_MODEL, (unsigned)a_rows, D_MODEL, BK, 128, 4, 3);
        tdm_load_2d_bf16(lds_addr_of(&Bs[cur ^ 1][0]), Wbf + (size_t)n0 * D_MODEL + kt,
                         D_MODEL, D_MODEL, D_MODEL, BK, 128, 4, 3);
        __builtin_amdgcn_s_wait_tensorcnt(2);   // only the 2 just-issued remain
      } else {
        __builtin_amdgcn_s_wait_tensorcnt(0);
      }
    }
    __syncthreads();

#pragma unroll
    for (int kk = 0; kk < 2; ++kk) {
      v16bf af[4], bfr[2];
#pragma unroll
      for (int mt = 0; mt < 4; ++mt)
        af[mt]  = load_frag_lds(As[cur], LDT, wm * 64 + mt * 16, kk * 32, lane);
#pragma unroll
      for (int nt = 0; nt < 2; ++nt)
        bfr[nt] = load_frag_lds(Bs[cur], LDT, wn * 32 + nt * 16, kk * 32, lane);
#pragma unroll
      for (int mt = 0; mt < 4; ++mt)
#pragma unroll
        for (int nt = 0; nt < 2; ++nt)
          acc[mt][nt] = WMMA_BF16(af[mt], bfr[nt], acc[mt][nt]);
    }
    __syncthreads();   // readers done before next prefetch overwrites
  }

  gemm_epilogue<MODE>(acc, bias, Outp, m0, n0, wm, wn, lane);
}

// ---------------------------------------------------------------------------
// GEMM fallback: f32 A/W converted while staging through LDS (BK=64).
// Used only if the workspace is too small for the bf16 mirrors.
// ---------------------------------------------------------------------------
template<int MODE, bool AF32>
__global__ __launch_bounds__(256)
void gemm_bf16_wmma(const void* __restrict__ Ap, const float* __restrict__ Wp,
                    const float* __restrict__ bias, void* __restrict__ Outp) {
  constexpr int BK = 64, LDT = 72;
  __shared__ __bf16 As[128 * LDT];
  __shared__ __bf16 Bs[128 * LDT];

  const int tid  = threadIdx.x;
  const int lane = tid & 31;
  const int wav  = tid >> 5;
  const int wm   = wav & 1;
  const int wn   = wav >> 1;
  const int m0   = blockIdx.x * 128;
  const int n0   = blockIdx.y * 128;

  v8f acc[4][2];
#pragma unroll
  for (int mt = 0; mt < 4; ++mt) { acc[mt][0] = vzero8(); acc[mt][1] = vzero8(); }

  const int sr = tid >> 3;         // 0..31
  const int sc = (tid & 7) * 8;    // 0..56

  for (int kt = 0; kt < D_MODEL; kt += BK) {
#pragma unroll
    for (int rr = 0; rr < 128; rr += 32) {
      const int row = sr + rr;
      if (AF32) {
        const float* gp = (const float*)Ap + (size_t)(m0 + row) * D_MODEL + kt + sc;
        *(v8bf*)&As[row * LDT + sc] = cvt_f32x8_bf16(gp);
      } else {
        const __bf16* gp = (const __bf16*)Ap + (size_t)(m0 + row) * D_MODEL + kt + sc;
        *(v8bf*)&As[row * LDT + sc] = *(const v8bf*)gp;
      }
      const float* wp = Wp + (size_t)(n0 + row) * D_MODEL + kt + sc;
      *(v8bf*)&Bs[row * LDT + sc] = cvt_f32x8_bf16(wp);
    }
    __syncthreads();

#pragma unroll
    for (int kk = 0; kk < 2; ++kk) {
      v16bf af[4], bfr[2];
#pragma unroll
      for (int mt = 0; mt < 4; ++mt)
        af[mt]  = load_frag_lds(As, LDT, wm * 64 + mt * 16, kk * 32, lane);
#pragma unroll
      for (int nt = 0; nt < 2; ++nt)
        bfr[nt] = load_frag_lds(Bs, LDT, wn * 32 + nt * 16, kk * 32, lane);
#pragma unroll
      for (int mt = 0; mt < 4; ++mt)
#pragma unroll
        for (int nt = 0; nt < 2; ++nt)
          acc[mt][nt] = WMMA_BF16(af[mt], bfr[nt], acc[mt][nt]);
    }
    __syncthreads();
  }

  gemm_epilogue<MODE>(acc, bias, Outp, m0, n0, wm, wn, lane);
}

// ---------------------------------------------------------------------------
// Flash attention: block = (b,h,128-row Q block), 8 waves x 16 rows.
// K/V 64x64 tiles staged by TDM (double buffered, shared by all 8 waves);
// TDM pads 32-dword rows with 4 dwords -> LDS stride 72 elems (conflict-free).
// Softmax row stats via DPP16 butterflies.
// ---------------------------------------------------------------------------
__global__ __launch_bounds__(256)
void flash_attn_wmma(const void* __restrict__ Qp, const void* __restrict__ Kp,
                     const void* __restrict__ Vp, void* __restrict__ Op) {
  constexpr int LDK = 72;
  const __bf16* Qh = (const __bf16*)Qp;
  const __bf16* Kh = (const __bf16*)Kp;
  const __bf16* Vt = (const __bf16*)Vp;
  __bf16* Attn = (__bf16*)Op;

  const int lane = threadIdx.x & 31;
  const int wave = threadIdx.x >> 5;   // 0..7
  const int bh   = blockIdx.x >> 4;    // 64 (b,h) pairs
  const int qblk = blockIdx.x & 15;    // 16 query blocks of 128
  const int b = bh >> 4, h = bh & 15;
  const int r = lane & 15, half = lane >> 4;

  const __bf16* Qb = Qh + (size_t)bh * SEQ * HD;
  const __bf16* Kb = Kh + (size_t)bh * SEQ * HD;
  const __bf16* Vb = Vt + (size_t)bh * HD * SEQ;

  const int q0 = qblk * 128 + wave * 16;

  __shared__ __bf16 Kls[2][64 * LDK];
  __shared__ __bf16 Vls[2][64 * LDK];
  __shared__ __bf16 pls[8][16][LDK];

  v16bf qf[2];
#pragma unroll
  for (int kk = 0; kk < 2; ++kk)
    qf[kk] = load_frag_ptr(Qb + (size_t)(q0 + r) * HD + kk * 32 + half * 8);

  float mrow[8], lrow[8];
  v8f o[4];
#pragma unroll
  for (int i = 0; i < 8; ++i) { mrow[i] = -3.0e38f; lrow[i] = 0.0f; }
#pragma unroll
  for (int nt = 0; nt < 4; ++nt) o[nt] = vzero8();

  constexpr float LOG2E = 1.4426950408889634f;
  constexpr float SCALE = 0.125f;      // 1/sqrt(64)
  constexpr int   NKB   = SEQ / 64;    // 32 key blocks

  if (wave == 0) {
    tdm_load_2d_bf16(lds_addr_of(&Kls[0][0]), Kb, HD, SEQ, HD, 64, 64, 4, 3);
    tdm_load_2d_bf16(lds_addr_of(&Vls[0][0]), Vb, SEQ, HD, SEQ, 64, 64, 4, 3);
  }

  for (int kb = 0; kb < NKB; ++kb) {
    const int cur = kb & 1;
    if (wave == 0) {
      if (kb + 1 < NKB) {
        const int key1 = (kb + 1) * 64;
        tdm_load_2d_bf16(lds_addr_of(&Kls[cur ^ 1][0]), Kb + (size_t)key1 * HD,
                         HD, SEQ, HD, 64, 64, 4, 3);
        tdm_load_2d_bf16(lds_addr_of(&Vls[cur ^ 1][0]), Vb + key1,
                         SEQ, HD, SEQ, 64, 64, 4, 3);
        __builtin_amdgcn_s_wait_tensorcnt(2);
      } else {
        __builtin_amdgcn_s_wait_tensorcnt(0);
      }
    }
    __syncthreads();

    // scores: S = (Q @ K^T) * scale, 16x64 per wave
    v8f sfr[4];
#pragma unroll
    for (int nt = 0; nt < 4; ++nt) {
      v8f z = vzero8();
#pragma unroll
      for (int kk = 0; kk < 2; ++kk) {
        v16bf kf = load_frag_ptr(&Kls[cur][(nt * 16 + r) * LDK + kk * 32 + half * 8]);
        z = WMMA_BF16(qf[kk], kf, z);
      }
#pragma unroll
      for (int i = 0; i < 8; ++i) z[i] *= SCALE;
      sfr[nt] = z;
    }

    // online softmax (row stats across the 16-lane half via DPP)
    float mnew[8], alpha[8], rsum[8];
#pragma unroll
    for (int i = 0; i < 8; ++i) {
      float v = fmaxf(fmaxf(sfr[0][i], sfr[1][i]), fmaxf(sfr[2][i], sfr[3][i]));
      v = red16_max(v);
      mnew[i]  = fmaxf(mrow[i], v);
      alpha[i] = exp2f((mrow[i] - mnew[i]) * LOG2E);
      mrow[i]  = mnew[i];
      rsum[i]  = 0.0f;
    }
#pragma unroll
    for (int nt = 0; nt < 4; ++nt)
#pragma unroll
      for (int i = 0; i < 8; ++i) {
        const float p = exp2f((sfr[nt][i] - mnew[i]) * LOG2E);
        sfr[nt][i] = p;
        rsum[i] += p;
      }
#pragma unroll
    for (int i = 0; i < 8; ++i)
      lrow[i] = lrow[i] * alpha[i] + red16_sum(rsum[i]);
#pragma unroll
    for (int nt = 0; nt < 4; ++nt)
#pragma unroll
      for (int i = 0; i < 8; ++i) o[nt][i] *= alpha[i];

    // C-layout -> A-layout for P via per-wave LDS tile (wave-local)
#pragma unroll
    for (int nt = 0; nt < 4; ++nt)
#pragma unroll
      for (int i = 0; i < 8; ++i)
        pls[wave][i + half * 8][nt * 16 + r] = (__bf16)sfr[nt][i];

    v16bf pf[2];
#pragma unroll
    for (int kk = 0; kk < 2; ++kk)
      pf[kk] = load_frag_ptr(&pls[wave][r][kk * 32 + half * 8]);

    // O += P @ V   (V^T tile: row = d, col = key)
#pragma unroll
    for (int nt = 0; nt < 4; ++nt)
#pragma unroll
      for (int kk = 0; kk < 2; ++kk) {
        v16bf vf = load_frag_ptr(&Vls[cur][(nt * 16 + r) * LDK + kk * 32 + half * 8]);
        o[nt] = WMMA_BF16(pf[kk], vf, o[nt]);
      }

    __syncthreads();   // all reads of Kls/Vls[cur] done before it is re-filled
  }

  float inv[8];
#pragma unroll
  for (int i = 0; i < 8; ++i) inv[i] = 1.0f / lrow[i];
#pragma unroll
  for (int nt = 0; nt < 4; ++nt)
#pragma unroll
    for (int i = 0; i < 8; ++i) {
      const int srow = q0 + i + half * 8;
      const int col  = h * 64 + nt * 16 + r;
      Attn[(size_t)(b * SEQ + srow) * D_MODEL + col] = (__bf16)(o[nt][i] * inv[i]);
    }
}

// ---------------------------------------------------------------------------
extern "C" void kernel_launch(void* const* d_in, const int* in_sizes, int n_in,
                              void* d_out, int out_size, void* d_ws, size_t ws_size,
                              hipStream_t stream) {
  (void)in_sizes; (void)n_in; (void)out_size;
  const float* q  = (const float*)d_in[0];
  const float* k  = (const float*)d_in[1];
  const float* v  = (const float*)d_in[2];
  const float* Wq = (const float*)d_in[3];
  const float* bq = (const float*)d_in[4];
  const float* Wk = (const float*)d_in[5];
  const float* bk = (const float*)d_in[6];
  const float* Wv = (const float*)d_in[7];
  const float* bv = (const float*)d_in[8];
  const float* Wo = (const float*)d_in[9];
  const float* bo = (const float*)d_in[10];

  char* ws = (char*)d_ws;
  const size_t SZ  = (size_t)MTOT * D_MODEL * 2;     // 16 MiB bf16 tensor
  const size_t WSZ = (size_t)D_MODEL * D_MODEL * 2;  //  2 MiB bf16 weight
  void* Qh = ws;
  void* Kh = ws + SZ;
  void* Vt = ws + 2 * SZ;
  void* At = ws + 3 * SZ;

  dim3 ggrid(MTOT / 128, D_MODEL / 128);
  const int fgrid = BATCH * HEADS * (SEQ / 128);   // 1024 blocks

  if (ws_size >= 7 * SZ + 4 * WSZ) {
    __bf16* Xq  = (__bf16*)(ws + 4 * SZ);
    __bf16* Xk  = (__bf16*)(ws + 5 * SZ);
    __bf16* Xv  = (__bf16*)(ws + 6 * SZ);
    __bf16* Wqb = (__bf16*)(ws + 7 * SZ);
    __bf16* Wkb = (__bf16*)(ws + 7 * SZ + WSZ);
    __bf16* Wvb = (__bf16*)(ws + 7 * SZ + 2 * WSZ);
    __bf16* Wob = (__bf16*)(ws + 7 * SZ + 3 * WSZ);

    const int nx8 = MTOT * D_MODEL / 8;        // 1048576
    const int nw8 = D_MODEL * D_MODEL / 8;     // 131072
    cvt_f32_to_bf16<<<nx8 / 256, 256, 0, stream>>>(q,  Xq,  nx8);
    cvt_f32_to_bf16<<<nx8 / 256, 256, 0, stream>>>(k,  Xk,  nx8);
    cvt_f32_to_bf16<<<nx8 / 256, 256, 0, stream>>>(v,  Xv,  nx8);
    cvt_f32_to_bf16<<<nw8 / 256, 256, 0, stream>>>(Wq, Wqb, nw8);
    cvt_f32_to_bf16<<<nw8 / 256, 256, 0, stream>>>(Wk, Wkb, nw8);
    cvt_f32_to_bf16<<<nw8 / 256, 256, 0, stream>>>(Wv, Wvb, nw8);
    cvt_f32_to_bf16<<<nw8 / 256, 256, 0, stream>>>(Wo, Wob, nw8);

    gemm_tdm_wmma<0><<<ggrid, 256, 0, stream>>>(Xq, Wqb, bq, Qh, MTOT);
    gemm_tdm_wmma<0><<<ggrid, 256, 0, stream>>>(Xk, Wkb, bk, Kh, MTOT);
    gemm_tdm_wmma<1><<<ggrid, 256, 0, stream>>>(Xv, Wvb, bv, Vt, MTOT);
    flash_attn_wmma<<<fgrid, 256, 0, stream>>>(Qh, Kh, Vt, At);
    gemm_tdm_wmma<2><<<ggrid, 256, 0, stream>>>((const __bf16*)At, Wob, bo, d_out, MTOT);
  } else {
    gemm_bf16_wmma<0, true ><<<ggrid, 256, 0, stream>>>(q,  Wq, bq, Qh);
    gemm_bf16_wmma<0, true ><<<ggrid, 256, 0, stream>>>(k,  Wk, bk, Kh);
    gemm_bf16_wmma<1, true ><<<ggrid, 256, 0, stream>>>(v,  Wv, bv, Vt);
    flash_attn_wmma<<<fgrid, 256, 0, stream>>>(Qh, Kh, Vt, At);
    gemm_bf16_wmma<2, false><<<ggrid, 256, 0, stream>>>(At, Wo, bo, d_out);
  }
}
